// KANGATConv_67482526154791
// MI455X (gfx1250) — matrix operand order repaired and checked by reading, not compiled, not run on hardware
//
#include <hip/hip_runtime.h>
#include <hip/hip_bf16.h>

// ---------------------------------------------------------------------------
// KAN-GAT forward for B=2, N=256, C=64, O=64, GRID=5, ORDER=3 (GK=8)
// ---------------------------------------------------------------------------

#define Bn  2
#define Nn  256
#define Cn  64
#define On  64
#define GKn 8

typedef __attribute__((ext_vector_type(16))) _Float16 v16h;
typedef __attribute__((ext_vector_type(8)))  float    v8f;

union HFrag { v16h v; _Float16 e[16]; uint4 q[2]; };
union FFrag { v8f  v; float     e[8];             };

// Uniform-knot cubic B-spline basis, 12 knots over [-2.2, 2.2], h = 0.4.
// Matches efficient-KAN Cox-de Boor with grid g[i] = (i-3)*0.4 - 1.
__device__ __forceinline__ void bspline8(float r, float* __restrict__ out) {
    const float h = 0.4f;
    float g[12];
#pragma unroll
    for (int i = 0; i < 12; ++i) g[i] = -2.2f + h * (float)i;
    float b[11];
#pragma unroll
    for (int i = 0; i < 11; ++i)
        b[i] = (r >= g[i] && r < g[i + 1]) ? 1.0f : 0.0f;
#pragma unroll
    for (int k = 1; k <= 3; ++k) {
        const float inv = 1.0f / (h * (float)k);
#pragma unroll
        for (int i = 0; i < 11 - k; ++i) {
            b[i] = (r - g[i]) * inv * b[i] + (g[i + k + 1] - r) * inv * b[i + 1];
        }
    }
#pragma unroll
    for (int i = 0; i < 8; ++i) out[i] = b[i];
}

__device__ __forceinline__ float silu(float v) {
    return v / (1.0f + __expf(-v));
}

// ---------------------------------------------------------------------------
// Kernel 1: energy[b,i,j] = KAN_f(x_i - x_j) (out_dim 1), mask by adj,
//           row softmax over j, write alpha as f16.  One block per (b,i).
// ---------------------------------------------------------------------------
__global__ __launch_bounds__(Nn)
void energy_softmax_kernel(const float* __restrict__ x,
                           const int* __restrict__ adj,
                           const float* __restrict__ fw_base,
                           const float* __restrict__ fw_spline,
                           const float* __restrict__ fw_scaler,
                           _Float16* __restrict__ alpha_h) {
    const int bi = blockIdx.x;          // b*N + i
    const int b  = bi / Nn;
    const int j  = threadIdx.x;         // 0..255

    __shared__ float s_xi[Cn];
    __shared__ float s_wb[Cn];
    __shared__ float s_ws[Cn * GKn];    // fw_spline * fw_scaler
    __shared__ float s_red[Nn];

    if (j < Cn) {
        s_xi[j] = x[(size_t)bi * Cn + j];
        s_wb[j] = fw_base[j];
    }
    for (int t = j; t < Cn * GKn; t += Nn) {
        s_ws[t] = fw_spline[t] * fw_scaler[t >> 3];
    }
    __syncthreads();

    const float* xj = x + ((size_t)b * Nn + j) * Cn;

    float e = 0.0f;
    for (int c = 0; c < Cn; ++c) {
        const float r = s_xi[c] - xj[c];
        float bs[GKn];
        bspline8(r, bs);
        float s = silu(r) * s_wb[c];
#pragma unroll
        for (int g = 0; g < GKn; ++g) s += bs[g] * s_ws[c * GKn + g];
        e += s;
    }
    if (adj[(size_t)bi * Nn + j] == 0) e = -1e9f;

    // --- row softmax (256 threads) ---
    s_red[j] = e;
    __syncthreads();
#pragma unroll
    for (int off = Nn / 2; off > 0; off >>= 1) {
        if (j < off) s_red[j] = fmaxf(s_red[j], s_red[j + off]);
        __syncthreads();
    }
    const float mx = s_red[0];
    __syncthreads();
    const float ex = __expf(e - mx);
    s_red[j] = ex;
    __syncthreads();
#pragma unroll
    for (int off = Nn / 2; off > 0; off >>= 1) {
        if (j < off) s_red[j] += s_red[j + off];
        __syncthreads();
    }
    const float a = ex / s_red[0];
    alpha_h[(size_t)bi * Nn + j] = (_Float16)a;
}

// ---------------------------------------------------------------------------
// Kernel 2: msg[b,n,o] = KAN_m(x[b,n,:]).  One block (64 threads) per row.
//           Output stored TRANSPOSED as f16: msg_t[b][o][j]  (WMMA B operand:
//           each WMMA lane then reads 16 contiguous f16 along K).
// ---------------------------------------------------------------------------
__global__ __launch_bounds__(On)
void kan_msg_kernel(const float* __restrict__ x,
                    const float* __restrict__ base_w,
                    const float* __restrict__ spline_w,
                    const float* __restrict__ scaler,
                    _Float16* __restrict__ msg_t) {
    const int row = blockIdx.x;         // b*N + n
    const int b   = row >> 8;
    const int n   = row & (Nn - 1);
    const int t   = threadIdx.x;        // 0..63

    __shared__ float s_silu[Cn];
    __shared__ float s_bs[Cn * GKn];

    {
        const float v = x[(size_t)row * Cn + t];
        s_silu[t] = silu(v);
        float bs[GKn];
        bspline8(v, bs);
#pragma unroll
        for (int g = 0; g < GKn; ++g) s_bs[t * GKn + g] = bs[g];
    }
    __syncthreads();

    const int o = t;
    float acc = 0.0f;
    for (int c = 0; c < Cn; ++c) {
        acc += s_silu[c] * base_w[o * Cn + c];
        const float* sw = spline_w + ((size_t)o * Cn + c) * GKn;
        float s2 = 0.0f;
#pragma unroll
        for (int g = 0; g < GKn; ++g) s2 += s_bs[c * GKn + g] * sw[g];
        acc += scaler[o * Cn + c] * s2;
    }
    msg_t[((size_t)b * On + o) * Nn + n] = (_Float16)acc;
}

// ---------------------------------------------------------------------------
// Kernel 3: aggr[b,i,o] = sum_j alpha[b,i,j] * msg[b,j,o]  via WMMA f16->f32.
// One wave per 16x64 output strip: A fragment shared across 4 N-tiles,
// 4 f32 accumulators, 32 WMMAs per wave.  All fragment loads are b128.
// ---------------------------------------------------------------------------
__global__ __launch_bounds__(32)
void aggr_wmma_kernel(const _Float16* __restrict__ alpha_h,
                      const _Float16* __restrict__ msg_t,
                      float* __restrict__ aggr) {
    const int blk = blockIdx.x;             // 32 blocks: b(2) x ti(16)
    const int b   = blk >> 4;
    const int ti  = blk & 15;

    const int lane = threadIdx.x;           // 0..31
    const int hi   = lane >> 4;             // half-wave select
    const int ln   = lane & 15;

    // A row for this lane (M = ti*16 + ln), contiguous along K
    const _Float16* Arow = alpha_h + ((size_t)b * Nn + ti * 16 + ln) * Nn;
    const _Float16* Bt   = msg_t   + (size_t)b * On * Nn;

    FFrag acc[4];
#pragma unroll
    for (int to = 0; to < 4; ++to) acc[to].v = (v8f){};

    for (int k0 = 0; k0 < Nn; k0 += 32) {
        // A (16x32 f16, MxK): e[i] = K = hi*16 + i
        HFrag a;
        a.q[0] = *(const uint4*)(Arow + k0 + hi * 16);
        a.q[1] = *(const uint4*)(Arow + k0 + hi * 16 + 8);

#pragma unroll
        for (int to = 0; to < 4; ++to) {
            const int n = to * 16 + ln;     // B column owned by this lane
            // B (32x16 f16, KxN): e[i] = msg[K = hi*16 + i][n] = msg_t[n][k0+hi*16+i]
            HFrag bf;
            const _Float16* bp = Bt + (size_t)n * Nn + k0 + hi * 16;
            bf.q[0] = *(const uint4*)(bp);
            bf.q[1] = *(const uint4*)(bp + 8);
            acc[to].v = __builtin_amdgcn_wmma_f32_16x16x32_f16(
                            false, a.v, false, bf.v, (short)0, acc[to].v,
                            false, false);
        }
    }

    // C/D 16x16 f32: VGPR p -> row (hi*8 + p), column ln
#pragma unroll
    for (int to = 0; to < 4; ++to) {
#pragma unroll
        for (int p = 0; p < 8; ++p) {
            const int mr = ti * 16 + hi * 8 + p;
            aggr[((size_t)b * Nn + mr) * On + to * 16 + ln] = acc[to].e[p];
        }
    }
}

// ---------------------------------------------------------------------------
// Kernel 4: out[b,n,:] = KAN_u([x[b,n,:] ; aggr[b,n,:]])  (in_dim = 128)
// ---------------------------------------------------------------------------
#define Un (Cn + On)   // 128

__global__ __launch_bounds__(On)
void kan_update_kernel(const float* __restrict__ x,
                       const float* __restrict__ aggr,
                       const float* __restrict__ base_w,
                       const float* __restrict__ spline_w,
                       const float* __restrict__ scaler,
                       float* __restrict__ out) {
    const int row = blockIdx.x;         // b*N + n
    const int t   = threadIdx.x;        // 0..63

    __shared__ float s_silu[Un];
    __shared__ float s_bs[Un * GKn];

#pragma unroll
    for (int rep = 0; rep < 2; ++rep) {
        const int c = rep * On + t;
        const float v = (c < Cn) ? x[(size_t)row * Cn + c]
                                 : aggr[(size_t)row * On + (c - Cn)];
        s_silu[c] = silu(v);
        float bs[GKn];
        bspline8(v, bs);
#pragma unroll
        for (int g = 0; g < GKn; ++g) s_bs[c * GKn + g] = bs[g];
    }
    __syncthreads();

    const int o = t;
    float acc = 0.0f;
    for (int c = 0; c < Un; ++c) {
        acc += s_silu[c] * base_w[o * Un + c];
        const float* sw = spline_w + ((size_t)o * Un + c) * GKn;
        float s2 = 0.0f;
#pragma unroll
        for (int g = 0; g < GKn; ++g) s2 += s_bs[c * GKn + g] * sw[g];
        acc += scaler[o * Un + c] * s2;
    }
    out[(size_t)row * On + o] = acc;
}

// ---------------------------------------------------------------------------
// Launch
// ---------------------------------------------------------------------------
extern "C" void kernel_launch(void* const* d_in, const int* in_sizes, int n_in,
                              void* d_out, int out_size, void* d_ws, size_t ws_size,
                              hipStream_t stream) {
    const float* x         = (const float*)d_in[0];
    const int*   adj       = (const int*)  d_in[1];
    const float* fw_base   = (const float*)d_in[2];
    const float* fw_spline = (const float*)d_in[3];
    const float* fw_scaler = (const float*)d_in[4];
    const float* mw_base   = (const float*)d_in[5];
    const float* mw_spline = (const float*)d_in[6];
    const float* mw_scaler = (const float*)d_in[7];
    const float* uw_base   = (const float*)d_in[8];
    const float* uw_spline = (const float*)d_in[9];
    const float* uw_scaler = (const float*)d_in[10];
    float* out = (float*)d_out;

    char* ws = (char*)d_ws;
    const size_t alpha_bytes = (size_t)Bn * Nn * Nn * sizeof(_Float16); // 256 KB
    const size_t msg_bytes   = (size_t)Bn * Nn * On * sizeof(_Float16); //  64 KB
    _Float16* alpha_h = (_Float16*)ws;
    _Float16* msg_t   = (_Float16*)(ws + alpha_bytes);
    float*    aggr    = (float*)   (ws + alpha_bytes + msg_bytes);

    energy_softmax_kernel<<<Bn * Nn, Nn, 0, stream>>>(
        x, adj, fw_base, fw_spline, fw_scaler, alpha_h);

    kan_msg_kernel<<<Bn * Nn, On, 0, stream>>>(
        x, mw_base, mw_spline, mw_scaler, msg_t);

    aggr_wmma_kernel<<<Bn * (Nn / 16), 32, 0, stream>>>(
        alpha_h, msg_t, aggr);

    kan_update_kernel<<<Bn * Nn, On, 0, stream>>>(
        x, aggr, uw_base, uw_spline, uw_scaler, out);
}
